// LSTMModel_72198400246317
// MI455X (gfx1250) — matrix operand order repaired and checked by reading, not compiled, not run on hardware
//
#include <hip/hip_runtime.h>
#include <hip/hip_bf16.h>
#include <math.h>

// ---------------------------------------------------------------------------
// Types for WMMA fragments (wave32, gfx1250)
// ---------------------------------------------------------------------------
typedef __attribute__((ext_vector_type(16))) _Float16 v16h;
typedef __attribute__((ext_vector_type(8)))  _Float16 v8h;
typedef __attribute__((ext_vector_type(8)))  float    v8f;

static __device__ __forceinline__ v16h cat8(v8h lo, v8h hi) {
  return __builtin_shufflevector(lo, hi, 0,1,2,3,4,5,6,7,8,9,10,11,12,13,14,15);
}

static __device__ __forceinline__ float sigmoidf_(float x) {
  return 1.0f / (1.0f + __expf(-x));
}

#define WMMA_F16(A, B, C) \
  __builtin_amdgcn_wmma_f32_16x16x32_f16(false, (A), false, (B), (short)0, (C), false, false)

// ---------------------------------------------------------------------------
// Utility kernels
// ---------------------------------------------------------------------------
__global__ void cvt_f32_to_f16(const float* __restrict__ in,
                               _Float16* __restrict__ out, int n) {
  int i = blockIdx.x * blockDim.x + threadIdx.x;
  if (i < n) out[i] = (_Float16)in[i];
}

// Build fused transposed weight Wt[4H, DinP+Hp] from k[Din,4H], r[H,4H].
// Row n (output column), col kk (K index). Zero-padded.
__global__ void build_wt(const float* __restrict__ K, const float* __restrict__ R,
                         _Float16* __restrict__ wt,
                         int Din, int DinP, int H, int Hp, int fourH) {
  const int Ktot = DinP + Hp;
  const size_t total = (size_t)fourH * (size_t)Ktot;
  for (size_t idx = (size_t)blockIdx.x * blockDim.x + threadIdx.x; idx < total;
       idx += (size_t)gridDim.x * blockDim.x) {
    int n  = (int)(idx / Ktot);
    int kk = (int)(idx % Ktot);
    float v = 0.0f;
    if (kk < Din) {
      v = K[(size_t)kk * fourH + n];
    } else {
      int k2 = kk - DinP;
      if (k2 >= 0 && k2 < H) v = R[(size_t)k2 * fourH + n];
    }
    wt[idx] = (_Float16)v;
  }
}

// ---------------------------------------------------------------------------
// Grid-wide barrier (persistent kernel, sense via generation counter)
// ---------------------------------------------------------------------------
static __device__ __forceinline__ void grid_sync(unsigned* bar, unsigned nBlocks) {
  __threadfence();      // make this thread's h/c stores device-visible
  __syncthreads();
  if (threadIdx.x == 0) {
    unsigned gen = __hip_atomic_load(&bar[1], __ATOMIC_RELAXED, __HIP_MEMORY_SCOPE_AGENT);
    unsigned arrived = __hip_atomic_fetch_add(&bar[0], 1u, __ATOMIC_ACQ_REL,
                                              __HIP_MEMORY_SCOPE_AGENT);
    if (arrived == nBlocks - 1u) {
      __hip_atomic_store(&bar[0], 0u, __ATOMIC_RELAXED, __HIP_MEMORY_SCOPE_AGENT);
      __hip_atomic_fetch_add(&bar[1], 1u, __ATOMIC_RELEASE, __HIP_MEMORY_SCOPE_AGENT);
    } else {
      while (__hip_atomic_load(&bar[1], __ATOMIC_ACQUIRE, __HIP_MEMORY_SCOPE_AGENT) == gen) {
        __builtin_amdgcn_s_sleep(1);
      }
    }
  }
  __syncthreads();
}

// ---------------------------------------------------------------------------
// One K-segment of the fused gate GEMM for an M=32 super-tile:
//   acc_g{0,1} += A{0,1}(16x32) * B_g(32x16),  g = i,f,c,o
// Each B fragment is loaded once and reused for both M-tiles (halves the
// dominant L2 weight stream vs one M-tile per wave).
// A fragment (16-bit A 16x32): lane m = lane&15, khalf = lane>>4,
//   elems 0..15 -> k = k0 + khalf*16 + e      (contiguous 32B = 2x16B)
// B fragment (16-bit B 32x16): lane n = lane&15, khalf = lane>>4,
//   elems 0..15 -> k = k0 + khalf*16 + e      (contiguous 32B = 2x16B)
// ---------------------------------------------------------------------------
static __device__ __forceinline__ void mma_seg2(
    const _Float16* __restrict__ aBase0, const _Float16* __restrict__ aBase1,
    int len, int khalf,
    const _Float16* __restrict__ wI, const _Float16* __restrict__ wF,
    const _Float16* __restrict__ wG, const _Float16* __restrict__ wO,
    v8f& accI0, v8f& accI1, v8f& accF0, v8f& accF1,
    v8f& accG0, v8f& accG1, v8f& accO0, v8f& accO1)
{
  for (int k0 = 0; k0 < len; k0 += 32) {
    const int ao = k0 + khalf * 16;
    v16h a0 = cat8(*(const v8h*)(aBase0 + ao), *(const v8h*)(aBase0 + ao + 8));
    v16h a1 = cat8(*(const v8h*)(aBase1 + ao), *(const v8h*)(aBase1 + ao + 8));
    const int bo = k0 + khalf * 16;
    v16h b;
    b = cat8(*(const v8h*)(wI + bo), *(const v8h*)(wI + bo + 8));
    accI0 = WMMA_F16(a0, b, accI0);
    accI1 = WMMA_F16(a1, b, accI1);
    b = cat8(*(const v8h*)(wF + bo), *(const v8h*)(wF + bo + 8));
    accF0 = WMMA_F16(a0, b, accF0);
    accF1 = WMMA_F16(a1, b, accF1);
    b = cat8(*(const v8h*)(wG + bo), *(const v8h*)(wG + bo + 8));
    accG0 = WMMA_F16(a0, b, accG0);
    accG1 = WMMA_F16(a1, b, accG1);
    b = cat8(*(const v8h*)(wO + bo), *(const v8h*)(wO + bo + 8));
    accO0 = WMMA_F16(a0, b, accO0);
    accO1 = WMMA_F16(a1, b, accO1);
  }
}

// ---------------------------------------------------------------------------
// Persistent LSTM scan. One wave owns one 32x16 super-tile of h (two M-tiles,
// one N-tile); it accumulates the 8 corresponding gate tiles of
// z = [x_t | h] @ Wt^T + b, then does the cell update locally.
// Ping-pong h buffers + grid barrier per timestep.
// Grid*blockDim/32 must equal 4 * (H/16) waves exactly.
// ---------------------------------------------------------------------------
__global__ void __launch_bounds__(256) lstm_scan(
    const _Float16* __restrict__ xseq,   // [128, T, DinP]
    const _Float16* __restrict__ wt,     // [4H, DinP+Hp]
    const float*    __restrict__ bias,   // [4H] gates i,f,g,o
    _Float16*       __restrict__ hstate, // [2, 128, Hp] (zeroed)
    float*          __restrict__ cstate, // [128, Hp]    (zeroed)
    _Float16*       __restrict__ outseq, // [128, T, Hp] or nullptr
    unsigned*       __restrict__ bar,    // [2] counter, generation (zeroed)
    int T, int DinP, int H, int Hp, int nColTiles, int nBlocks)
{
  const int Ktot  = DinP + Hp;
  const int lane  = threadIdx.x & 31;
  const int ln    = lane & 15;
  const int khalf = lane >> 4;
  const int wave  = blockIdx.x * (blockDim.x >> 5) + (threadIdx.x >> 5);

  const int colTile = wave % nColTiles;
  const int rowSup  = wave / nColTiles;   // 0..3 (B=128 -> 4 super row tiles of 32)
  const int j0 = colTile * 16;
  const int m0 = rowSup * 32;

  // Per-lane bias (one column per lane; broadcast across the 8 acc elements)
  const float bI = bias[0 * H + j0 + ln];
  const float bF = bias[1 * H + j0 + ln];
  const float bG = bias[2 * H + j0 + ln];
  const float bO = bias[3 * H + j0 + ln];

  // Per-lane weight row pointers (row = output column)
  const _Float16* wIrow = wt + (size_t)(0 * H + j0 + ln) * Ktot;
  const _Float16* wFrow = wt + (size_t)(1 * H + j0 + ln) * Ktot;
  const _Float16* wGrow = wt + (size_t)(2 * H + j0 + ln) * Ktot;
  const _Float16* wOrow = wt + (size_t)(3 * H + j0 + ln) * Ktot;

  // Per-lane A row bases (batch rows m0+ln and m0+16+ln)
  const _Float16* xrow0 = xseq + (size_t)(m0 + ln) * T * DinP;
  const _Float16* xrow1 = xseq + (size_t)(m0 + 16 + ln) * T * DinP;
  const size_t hrowOff0 = (size_t)(m0 + ln) * Hp;
  const size_t hrowOff1 = (size_t)(m0 + 16 + ln) * Hp;
  const size_t hbufSz   = (size_t)128 * Hp;

  for (int t = 0; t < T; ++t) {
    const int rp = t & 1;
    const int wp = rp ^ 1;

    v8f accI0 = { bI, bI, bI, bI, bI, bI, bI, bI };
    v8f accF0 = { bF, bF, bF, bF, bF, bF, bF, bF };
    v8f accG0 = { bG, bG, bG, bG, bG, bG, bG, bG };
    v8f accO0 = { bO, bO, bO, bO, bO, bO, bO, bO };
    v8f accI1 = accI0, accF1 = accF0, accG1 = accG0, accO1 = accO0;

    // Segment 1: input x_t (K in [0, DinP))
    mma_seg2(xrow0 + (size_t)t * DinP, xrow1 + (size_t)t * DinP, DinP, khalf,
             wIrow, wFrow, wGrow, wOrow,
             accI0, accI1, accF0, accF1, accG0, accG1, accO0, accO1);
    // Segment 2: recurrent h (K in [DinP, DinP+Hp))
    const _Float16* hbase = hstate + (size_t)rp * hbufSz;
    mma_seg2(hbase + hrowOff0, hbase + hrowOff1, Hp, khalf,
             wIrow + DinP, wFrow + DinP, wGrow + DinP, wOrow + DinP,
             accI0, accI1, accF0, accF1, accG0, accG1, accO0, accO1);

    // Pointwise cell update. C/D layout: lanes 0-15 N=lane, M=e; lanes 16-31
    // N=lane-16, M=8+e  ->  m = m0 + r*16 + khalf*8 + e, j = j0 + ln.
    #pragma unroll
    for (int r = 0; r < 2; ++r) {
      const v8f& aI = r ? accI1 : accI0;
      const v8f& aF = r ? accF1 : accF0;
      const v8f& aG = r ? accG1 : accG0;
      const v8f& aO = r ? accO1 : accO0;
      #pragma unroll
      for (int e = 0; e < 8; ++e) {
        const int m = m0 + r * 16 + khalf * 8 + e;
        const size_t cidx = (size_t)m * Hp + (j0 + ln);
        float ig = sigmoidf_(aI[e]);
        float fg = sigmoidf_(aF[e]);
        float gg = tanhf(aG[e]);
        float og = sigmoidf_(aO[e]);
        float c  = fg * cstate[cidx] + ig * gg;
        cstate[cidx] = c;
        float h = og * tanhf(c);
        _Float16 hh = (_Float16)h;
        hstate[(size_t)wp * hbufSz + cidx] = hh;
        if (outseq) outseq[((size_t)m * T + t) * Hp + (j0 + ln)] = hh;
      }
    }

    grid_sync(bar, (unsigned)nBlocks);
  }
}

// ---------------------------------------------------------------------------
// Dense head: one block (64 threads) per batch row; LDS ping-pong; softmax.
// Dims: 256 -> 512 -> 256 -> 128 -> 64 -> 16 -> 3
// ---------------------------------------------------------------------------
static __device__ __forceinline__ void dense_layer(
    const float* __restrict__ w, const float* __restrict__ b,
    int din, int dout, const float* __restrict__ src, float* __restrict__ dst,
    bool relu)
{
  for (int j = threadIdx.x; j < dout; j += 64) {
    float s = b[j];
    for (int i = 0; i < din; ++i) s += src[i] * w[(size_t)i * dout + j];
    dst[j] = relu ? fmaxf(s, 0.0f) : s;
  }
  __syncthreads();
}

__global__ void __launch_bounds__(64) dense_head(
    const _Float16* __restrict__ hfin,  // [128, 256] (final h of layer 4)
    const float* w1, const float* b1, const float* w2, const float* b2,
    const float* w3, const float* b3, const float* w4, const float* b4,
    const float* w5, const float* b5, const float* w6, const float* b6,
    float* __restrict__ out)            // [128, 3]
{
  __shared__ float bufA[512];
  __shared__ float bufB[512];
  const int b = blockIdx.x;
  for (int i = threadIdx.x; i < 256; i += 64)
    bufA[i] = (float)hfin[(size_t)b * 256 + i];
  __syncthreads();

  dense_layer(w1, b1, 256, 512, bufA, bufB, true);
  dense_layer(w2, b2, 512, 256, bufB, bufA, true);
  dense_layer(w3, b3, 256, 128, bufA, bufB, true);
  dense_layer(w4, b4, 128,  64, bufB, bufA, true);
  dense_layer(w5, b5,  64,  16, bufA, bufB, true);
  dense_layer(w6, b6,  16,   3, bufB, bufA, false);

  if (threadIdx.x == 0) {
    float m = fmaxf(bufA[0], fmaxf(bufA[1], bufA[2]));
    float e0 = __expf(bufA[0] - m);
    float e1 = __expf(bufA[1] - m);
    float e2 = __expf(bufA[2] - m);
    float s = e0 + e1 + e2;
    out[(size_t)b * 3 + 0] = e0 / s;
    out[(size_t)b * 3 + 1] = e1 / s;
    out[(size_t)b * 3 + 2] = e2 / s;
  }
}

// ---------------------------------------------------------------------------
// Host launcher
// ---------------------------------------------------------------------------
extern "C" void kernel_launch(void* const* d_in, const int* in_sizes, int n_in,
                              void* d_out, int out_size, void* d_ws, size_t ws_size,
                              hipStream_t stream) {
  (void)in_sizes; (void)n_in; (void)out_size; (void)ws_size;
  const int B = 128, T = 512;

  const float* x  = (const float*)d_in[0];
  const float* K1 = (const float*)d_in[1];  const float* R1 = (const float*)d_in[2];  const float* Bi1 = (const float*)d_in[3];
  const float* K2 = (const float*)d_in[4];  const float* R2 = (const float*)d_in[5];  const float* Bi2 = (const float*)d_in[6];
  const float* K3 = (const float*)d_in[7];  const float* R3 = (const float*)d_in[8];  const float* Bi3 = (const float*)d_in[9];
  const float* K4 = (const float*)d_in[10]; const float* R4 = (const float*)d_in[11]; const float* Bi4 = (const float*)d_in[12];
  const float* wd1 = (const float*)d_in[13]; const float* bd1 = (const float*)d_in[14];
  const float* wd2 = (const float*)d_in[15]; const float* bd2 = (const float*)d_in[16];
  const float* wd3 = (const float*)d_in[17]; const float* bd3 = (const float*)d_in[18];
  const float* wd4 = (const float*)d_in[19]; const float* bd4 = (const float*)d_in[20];
  const float* wd5 = (const float*)d_in[21]; const float* bd5 = (const float*)d_in[22];
  const float* wd6 = (const float*)d_in[23]; const float* bd6 = (const float*)d_in[24];
  float* out = (float*)d_out;

  // Per-layer geometry: Din (real), DinP (pad32), H, Hp (pad32)
  const int Din[4]  = { 64, 640, 400, 256 };
  const int DinP[4] = { 64, 640, 416, 256 };
  const int Hh[4]   = { 640, 400, 256, 256 };
  const int Hp[4]   = { 640, 416, 256, 256 };

  // Exact wave coverage: waves = 4 * (H/16) -> (blocks, threads) per layer
  const int scanBlocks[4]  = { 20, 25, 8, 8 };
  const int scanThreads[4] = { 256, 128, 256, 256 };

  // Workspace carving (256B aligned)
  char* p = (char*)d_ws;
  size_t used = 0;
  auto carve = [&](size_t bytes) -> char* {
    char* r = p + used;
    used += (bytes + 255) & ~(size_t)255;
    return r;
  };
  _Float16* xseq0 = (_Float16*)carve((size_t)B * T * 64 * 2);
  _Float16* seqBuf[4];
  _Float16* wtBuf[4];
  _Float16* hBuf[4];
  float*    cBuf[4];
  for (int l = 0; l < 4; ++l) {
    seqBuf[l] = (l < 3) ? (_Float16*)carve((size_t)B * T * Hp[l] * 2) : nullptr;
    wtBuf[l]  = (_Float16*)carve((size_t)4 * Hh[l] * (DinP[l] + Hp[l]) * 2);
    hBuf[l]   = (_Float16*)carve((size_t)2 * B * Hp[l] * 2);
    cBuf[l]   = (float*)carve((size_t)B * Hp[l] * 4);
  }
  unsigned* bar = (unsigned*)carve(256);

  // Zero everything we use (h/c/barrier + zero-padding; deterministic per call)
  hipMemsetAsync(d_ws, 0, used, stream);

  // Convert input x to f16
  {
    int n = B * T * 64;
    cvt_f32_to_f16<<<(n + 255) / 256, 256, 0, stream>>>(x, xseq0, n);
  }
  // Build fused transposed weights per layer
  const float* Ks[4] = { K1, K2, K3, K4 };
  const float* Rs[4] = { R1, R2, R3, R4 };
  for (int l = 0; l < 4; ++l) {
    build_wt<<<1024, 256, 0, stream>>>(Ks[l], Rs[l], wtBuf[l],
                                       Din[l], DinP[l], Hh[l], Hp[l], 4 * Hh[l]);
  }

  // LSTM scans
  const float* Bs[4] = { Bi1, Bi2, Bi3, Bi4 };
  const _Float16* inSeq[4] = { xseq0, seqBuf[0], seqBuf[1], seqBuf[2] };
  for (int l = 0; l < 4; ++l) {
    int nColTiles = Hh[l] / 16;
    lstm_scan<<<scanBlocks[l], scanThreads[l], 0, stream>>>(
        inSeq[l], wtBuf[l], Bs[l], hBuf[l], cBuf[l], seqBuf[l], bar,
        T, DinP[l], Hh[l], Hp[l], nColTiles, scanBlocks[l]);
  }

  // Final h of layer 4: last step t=511 writes ping-pong buffer 0
  dense_head<<<B, 64, 0, stream>>>(hBuf[3],
                                   wd1, bd1, wd2, bd2, wd3, bd3,
                                   wd4, bd4, wd5, bd5, wd6, bd6, out);
}
